// LatencyPredictor_6579889897816
// MI455X (gfx1250) — matrix operand order repaired and checked by reading, not compiled, not run on hardware
//
#include <hip/hip_runtime.h>

typedef __attribute__((ext_vector_type(16))) __bf16 v16bf;
typedef __attribute__((ext_vector_type(8)))  float  v8f;

constexpr int Bn = 256, Ln = 2048, Tn = 64, K1 = 32, K2 = 16;
constexpr int LP    = Ln - Tn + 1;        // 1985
constexpr int TILES = (LP + 15) / 16;     // 125 (last tile: 1 valid row)
constexpr long long DA = (long long)Bn * LP * K2;   // d2 / one-hot size
constexpr long long SC = (long long)Bn * LP;        // per-step scalar size
constexpr long long PS = 2 * DA + 4 * SC + Bn;      // per-stream output stride

// ---------------------------------------------------------------------------
// Kernel 1: fused hankel-conv -> WMMA(bf16) -> relu -> WMMA(bf16) -> relu
//           -> max/argmax -> write d2, idx, vals, one-hot, soma input (ws)
// One wave handles 16 consecutive l positions of one (stream, batch) row.
// ---------------------------------------------------------------------------
__global__ __launch_bounds__(128)
void dendrite_kernel(const float* __restrict__ xa, const float* __restrict__ xb,
                     const float* __restrict__ W1a, const float* __restrict__ b1a,
                     const float* __restrict__ W2a, const float* __restrict__ b2a,
                     const float* __restrict__ W1b, const float* __restrict__ b1b,
                     const float* __restrict__ W2b, const float* __restrict__ b2b,
                     float* __restrict__ out, float* __restrict__ Iws)
{
    __shared__ float ld1[4][16 * 32];   // per-wave d1 staging (f32)
    __shared__ float ld2[4][16 * 16];   // per-wave d2 staging (f32)

    const int lane = threadIdx.x & 31;
    const int wib  = threadIdx.x >> 5;
    const int gw   = blockIdx.x * 4 + wib;          // 0 .. 2*Bn*TILES-1
    const int strm = gw / (Bn * TILES);
    const int rem  = gw % (Bn * TILES);
    const int b    = rem / TILES;
    const int tile = rem % TILES;
    const int l0   = tile * 16;

    const float* x  = strm ? xb  : xa;
    const float* W1 = strm ? W1b : W1a;
    const float* b1 = strm ? b1b : b1a;
    const float* W2 = strm ? W2b : W2a;
    const float* b2 = strm ? b2b : b2a;

    const int nIdx = lane & 15;   // column (N) within 16-wide tile
    const int hi   = lane >> 4;   // lane-half selects K range
    const int m    = lane & 15;   // A-matrix row within tile
    const int k8   = hi * 8;      // A-matrix K sub-offset

    // ---- B operands: W1^T as two n-tiles x two k-chunks (32x16 bf16 layout)
    // B[k][n] = W1[n][k]  -> lane reads two 8-element runs of a W1 row.
    v16bf bW1[2][2];
    #pragma unroll
    for (int nt = 0; nt < 2; ++nt)
        #pragma unroll
        for (int kc = 0; kc < 2; ++kc) {
            const float* src = W1 + (nt * 16 + nIdx) * Tn + kc * 32 + hi * 8;
            #pragma unroll
            for (int j = 0; j < 16; ++j) bW1[nt][kc][j] = (__bf16)src[j < 8 ? j : j + 8];
        }

    // ---- A operand: 16 overlapping windows of x (16x32 bf16 layout x2 chunks)
    const int  ml   = min(l0 + m, LP - 1);          // clamp rows on edge tile
    const float* xrow = x + (long long)b * Ln + ml;
    __builtin_prefetch(xrow, 0, 0);                 // global_prefetch_b8
    v16bf aX[2];
    #pragma unroll
    for (int kc = 0; kc < 2; ++kc) {
        const float* src = xrow + kc * 32 + k8;
        #pragma unroll
        for (int j = 0; j < 8; ++j) {
            aX[kc][j]     = (__bf16)src[j];         // K = k8 + j
            aX[kc][8 + j] = (__bf16)src[16 + j];    // K = 16 + k8 + j
        }
    }

    // ---- d1 = relu(H @ W1^T + b1): two 16-wide n-tiles, K=64 in 2 steps ----
    #pragma unroll
    for (int nt = 0; nt < 2; ++nt) {
        v8f acc = {};
        acc = __builtin_amdgcn_wmma_f32_16x16x32_bf16(false, aX[0], false, bW1[nt][0],
                                                      (short)0, acc, false, false);
        acc = __builtin_amdgcn_wmma_f32_16x16x32_bf16(false, aX[1], false, bW1[nt][1],
                                                      (short)0, acc, false, false);
        const float bias = b1[nt * 16 + nIdx];
        #pragma unroll
        for (int r = 0; r < 8; ++r) {
            float v = acc[r] + bias;
            v = v > 0.f ? v : 0.f;
            ld1[wib][(r + hi * 8) * 32 + nt * 16 + nIdx] = v;  // row-major [16][32]
        }
    }
    __syncthreads();

    // ---- repack d1 as bf16 A operand (16x32) ----
    v16bf aD1;
    #pragma unroll
    for (int j = 0; j < 8; ++j) {
        aD1[j]     = (__bf16)ld1[wib][m * 32 + k8 + j];
        aD1[8 + j] = (__bf16)ld1[wib][m * 32 + 16 + k8 + j];
    }

    // ---- B operand: W2^T (B[k1][j] = W2[j][k1]) ----
    v16bf bW2;
    {
        const float* src = W2 + nIdx * K1 + hi * 8;
        #pragma unroll
        for (int j = 0; j < 16; ++j) bW2[j] = (__bf16)src[j < 8 ? j : j + 8];
    }

    // ---- d2 = relu(d1 @ W2^T + b2) ----
    v8f accd2 = {};
    accd2 = __builtin_amdgcn_wmma_f32_16x16x32_bf16(false, aD1, false, bW2,
                                                    (short)0, accd2, false, false);
    {
        const float bias = b2[nIdx];
        #pragma unroll
        for (int r = 0; r < 8; ++r) {
            float v = accd2[r] + bias;
            v = v > 0.f ? v : 0.f;
            ld2[wib][(r + hi * 8) * 16 + nIdx] = v;
        }
    }
    __syncthreads();

    // ---- per-row epilogue: lanes 0..15 each own one l position ----
    float* outS = out + (long long)strm * PS;
    float* da   = outS;
    float* ida  = outS + DA;
    float* vala = outS + DA + SC;
    float* wa   = outS + DA + 2 * SC;
    if (lane < 16) {
        const int l = l0 + lane;
        if (l < LP) {
            const float* row = &ld2[wib][lane * 16];
            float mx = row[0];
            int   mi = 0;
            #pragma unroll
            for (int k = 1; k < K2; ++k) {
                float v = row[k];
                if (v > mx) { mx = v; mi = k; }     // first max (ties) like argmax
            }
            const long long rowOff = ((long long)b * LP + l) * K2;
            #pragma unroll
            for (int k = 0; k < K2; k += 4) {
                *(float4*)(da + rowOff + k) = *(const float4*)(row + k);
                *(float4*)(wa + rowOff + k) = make_float4(k     == mi ? 1.f : 0.f,
                                                          k + 1 == mi ? 1.f : 0.f,
                                                          k + 2 == mi ? 1.f : 0.f,
                                                          k + 3 == mi ? 1.f : 0.f);
            }
            const long long sIdx = (long long)b * LP + l;
            ida[sIdx]  = (float)mi;
            vala[sIdx] = mx;                         // vals == soma_in == max_k d2
            Iws[(long long)strm * SC + sIdx] = mx;   // stage LIF input
        }
    }
}

// ---------------------------------------------------------------------------
// Kernel 2: leaky integrate-and-fire (hard forward). One wave per (stream,b)
// row: 32 coalesced loads per chunk, scalar recurrence via shuffle broadcast.
// ---------------------------------------------------------------------------
__global__ __launch_bounds__(128)
void lif_kernel(const float* __restrict__ Iws, float* __restrict__ out)
{
    const int lane = threadIdx.x & 31;
    const int wib  = threadIdx.x >> 5;
    const int gw   = blockIdx.x * 4 + wib;          // 0 .. 511
    const int strm = gw >> 8;
    const int b    = gw & 255;

    const float* I = Iws + (long long)strm * SC + (long long)b * LP;
    float* outS   = out  + (long long)strm * PS;
    float* sa     = outS + 2 * DA + 2 * SC + (long long)b * LP;
    float* series = outS + 2 * DA + 3 * SC + (long long)b * LP;
    float* first  = outS + 2 * DA + 4 * SC;

    const float decay = 1.f - 1.f / 20.f;           // TAU = 20
    float v = 0.f, cum = 0.f;
    int firstIdx = -1;

    for (int c0 = 0; c0 < LP; c0 += 32) {
        const int t  = c0 + lane;
        const float Il = (t < LP) ? I[t] : 0.f;     // coalesced chunk load
        float myS = 0.f, myCum = 0.f;
        const int nstep = min(32, LP - c0);
        for (int j = 0; j < nstep; ++j) {
            const float Ij = __shfl(Il, j, 32);     // broadcast step input
            v = v * decay + Ij;
            const float s = (v >= 1.f) ? 1.f : 0.f; // TH=1, forward = hard spike
            cum += s;
            if (s > 0.f) { if (firstIdx < 0) firstIdx = c0 + j; v = 0.f; }
            if (lane == j) { myS = s; myCum = cum; }
        }
        if (t < LP) { sa[t] = myS; series[t] = myCum; }
    }
    if (lane == 0) first[b] = (float)(firstIdx < 0 ? LP : firstIdx);
}

// ---------------------------------------------------------------------------
extern "C" void kernel_launch(void* const* d_in, const int* in_sizes, int n_in,
                              void* d_out, int out_size, void* d_ws, size_t ws_size,
                              hipStream_t stream)
{
    const float* xa  = (const float*)d_in[0];
    const float* xb  = (const float*)d_in[1];
    const float* W1a = (const float*)d_in[2];
    const float* b1a = (const float*)d_in[3];
    const float* W2a = (const float*)d_in[4];
    const float* b2a = (const float*)d_in[5];
    const float* W1b = (const float*)d_in[6];
    const float* b1b = (const float*)d_in[7];
    const float* W2b = (const float*)d_in[8];
    const float* b2b = (const float*)d_in[9];
    float* out = (float*)d_out;
    float* Iws = (float*)d_ws;                      // 2*SC floats ≈ 4 MB

    const int waves1  = 2 * Bn * TILES;             // 64000 waves, 4 per block
    dendrite_kernel<<<waves1 / 4, 128, 0, stream>>>(xa, xb, W1a, b1a, W2a, b2a,
                                                    W1b, b1b, W2b, b2b, out, Iws);
    lif_kernel<<<(2 * Bn) / 4, 128, 0, stream>>>(Iws, out);
}
